// RGCN2_53833120088190
// MI455X (gfx1250) — matrix operand order repaired and checked by reading, not compiled, not run on hardware
//
#include <hip/hip_runtime.h>

#define N_NODES   50000
#define N_EDGES   600000
#define N_FEAT    128
#define HIDDEN    128
#define N_REL     8
#define N_CLASSES 10

typedef _Float16 v16h __attribute__((ext_vector_type(16)));
typedef _Float16 v8h  __attribute__((ext_vector_type(8)));
typedef float    v8f  __attribute__((ext_vector_type(8)));

// ---------------------------------------------------------------- utility
__global__ void zero_kernel(float* __restrict__ p, int n) {
    int i = blockIdx.x * blockDim.x + threadIdx.x;
    if (i < n) p[i] = 0.0f;
}

// f32 -> f16 elementwise cast
__global__ void cast_kernel(const float* __restrict__ in, _Float16* __restrict__ out, int n) {
    int i = blockIdx.x * blockDim.x + threadIdx.x;
    if (i < n) out[i] = (_Float16)in[i];
}

// Wt[mat][col][k] = (f16) W[mat][k][col]  for 128x128 matrices (gridDim.y = mat)
__global__ void transpose_cast_kernel(const float* __restrict__ W, _Float16* __restrict__ Wt) {
    int mat = blockIdx.y;
    int col = blockIdx.x;
    int k   = threadIdx.x;
    Wt[((size_t)mat * 128 + col) * 128 + k] = (_Float16)W[((size_t)mat * 128 + k) * 128 + col];
}

// count edges per (dst, relation) -> float counts
__global__ void count_kernel(const int* __restrict__ dst, const int* __restrict__ et,
                             float* __restrict__ cnt) {
    int e = blockIdx.x * blockDim.x + threadIdx.x;
    if (e < N_EDGES) atomicAdd(&cnt[dst[e] * N_REL + et[e]], 1.0f);
}

// cnt -> 1/cnt (0 where empty), matching reference's where(cnt>0, 1/max(cnt,1), 0)
__global__ void inv_kernel(float* __restrict__ cnt) {
    int i = blockIdx.x * blockDim.x + threadIdx.x;
    if (i < N_NODES * N_REL) {
        float c = cnt[i];
        cnt[i] = (c > 0.0f) ? (1.0f / c) : 0.0f;
    }
}

// ---------------------------------------------------------------- WMMA GEMM core
// One wave computes a 16(M) x 128(N) strip of  Xh[row0:row0+16, :] @ Wt^T  where
// Xh is [*,128] f16 row-major and Wt is [128(col),128(k)] f16 (pre-transposed weights).
// Fragment layouts used here (wave32):
//   A (f16 16x32):  M = lane%16, K = j + 16*(lane>=16)   (j = half index 0..15)
//   B (f16 32x16):  N = lane%16, K = j + 16*(lane>=16)
//   C/D (f32 16x16): M = i + 8*(lane>=16), N = lane%16             (i = VGPR index 0..7)
// All 9 fragment loads of a k-step are issued as one clause, then a sched_barrier
// pins the 8 WMMAs after them: one s_wait per k-step instead of one per WMMA, and
// the next k-step's loads may still be hoisted into the WMMA shadow.
__device__ __forceinline__ void gemm16x128(const _Float16* __restrict__ xh,
                                           const _Float16* __restrict__ Wt,
                                           int row0, int lane, v8f acc[8]) {
    const int half = (lane >= 16) ? 1 : 0;
    const int mA   = lane & 15;
    const int nB   = lane & 15;
    const _Float16* wcol = Wt + (size_t)nB * 128 + half * 16;   // + nt*16*128 + k0
    #pragma unroll
    for (int k0 = 0; k0 < 128; k0 += 32) {
        // A fragment: one contiguous 16-half run (32B aligned), K = j + 16*(lane>=16)
        const _Float16* ar = xh + (size_t)(row0 + mA) * 128 + k0 + half * 16;
        v16h a = *(const v16h*)(ar);
        // batch-load all 8 B fragments (16 x b128 in one clause)
        v16h bfrag[8];
        #pragma unroll
        for (int nt = 0; nt < 8; ++nt)
            bfrag[nt] = *(const v16h*)(wcol + (size_t)nt * 16 * 128 + k0);
        // keep loads above, WMMAs below
        __builtin_amdgcn_sched_barrier(0);
        #pragma unroll
        for (int nt = 0; nt < 8; ++nt)
            acc[nt] = __builtin_amdgcn_wmma_f32_16x16x32_f16(
                false, a, false, bfrag[nt], (short)0, acc[nt], false, false);
    }
}

// h[n, rel, :] = x[n, :] @ W[rel]     (grid.y = relation; Wt pre-transposed f16)
__global__ void __launch_bounds__(128) transform_kernel(const _Float16* __restrict__ xh,
                                                        const _Float16* __restrict__ Wt,
                                                        float* __restrict__ h) {
    const int lane  = threadIdx.x & 31;
    const int wave  = threadIdx.x >> 5;
    const int mtile = blockIdx.x * 4 + wave;
    if (mtile >= N_NODES / 16) return;            // wave-uniform: EXEC stays all-ones
    const int rel  = blockIdx.y;
    const int row0 = mtile * 16;

    v8f acc[8] = {};
    gemm16x128(xh, Wt + (size_t)rel * N_FEAT * HIDDEN, row0, lane, acc);

    const int half = (lane >= 16) ? 1 : 0;
    const int nB   = lane & 15;
    const int mC   = half * 8;
    #pragma unroll
    for (int nt = 0; nt < 8; ++nt) {
        #pragma unroll
        for (int i = 0; i < 8; ++i) {
            int row = row0 + mC + i;
            h[(size_t)row * (N_REL * HIDDEN) + rel * HIDDEN + nt * 16 + nB] = acc[nt][i];
        }
    }
}

// out[n,:] = relu(agg[n,:] + xin[n,:] @ root + bias)   (rootT pre-transposed f16)
__global__ void __launch_bounds__(128) root_relu_kernel(const _Float16* __restrict__ xh,
                                                        const _Float16* __restrict__ rootT,
                                                        const float* __restrict__ bias,
                                                        const float* __restrict__ agg,
                                                        float* __restrict__ out) {
    const int lane  = threadIdx.x & 31;
    const int wave  = threadIdx.x >> 5;
    const int mtile = blockIdx.x * 4 + wave;
    if (mtile >= N_NODES / 16) return;
    const int row0 = mtile * 16;

    v8f acc[8] = {};
    gemm16x128(xh, rootT, row0, lane, acc);

    const int half = (lane >= 16) ? 1 : 0;
    const int nB   = lane & 15;
    const int mC   = half * 8;
    #pragma unroll
    for (int nt = 0; nt < 8; ++nt) {
        #pragma unroll
        for (int i = 0; i < 8; ++i) {
            int row = row0 + mC + i;
            int col = nt * 16 + nB;
            float v = acc[nt][i] + agg[(size_t)row * HIDDEN + col] + bias[col];
            out[(size_t)row * HIDDEN + col] = fmaxf(v, 0.0f);
        }
    }
}

// per edge: agg[dst,:] += h[src, etype, :] * inv[dst, etype]   (2 edges / 256-thread block)
__global__ void __launch_bounds__(256) edge_msg_kernel(const float* __restrict__ h,
                                                       const int* __restrict__ src,
                                                       const int* __restrict__ dst,
                                                       const int* __restrict__ et,
                                                       const float* __restrict__ inv,
                                                       float* __restrict__ agg) {
    int e = blockIdx.x * 2 + (threadIdx.x >> 7);
    int t = threadIdx.x & 127;
    if (e >= N_EDGES) return;
    int s = src[e], d = dst[e], r = et[e];
    float w = inv[(size_t)d * N_REL + r];
    float v = h[(size_t)s * (N_REL * HIDDEN) + (size_t)r * HIDDEN + t] * w;
    atomicAdd(&agg[(size_t)d * HIDDEN + t], v);
}

// out[n, c] = concat(out1[n], out2[n]) @ lin_w + lin_b    (one wave per node)
__global__ void __launch_bounds__(256) final_kernel(const float* __restrict__ o1,
                                                    const float* __restrict__ o2,
                                                    const float* __restrict__ lw,
                                                    const float* __restrict__ lb,
                                                    float* __restrict__ out) {
    int wave = threadIdx.x >> 5, lane = threadIdx.x & 31;
    int n = blockIdx.x * 8 + wave;
    if (n >= N_NODES) return;
    float p[N_CLASSES];
    #pragma unroll
    for (int c = 0; c < N_CLASSES; ++c) p[c] = 0.0f;
    #pragma unroll
    for (int j = 0; j < 4; ++j) {
        int k = lane + j * 32;
        float v1 = o1[(size_t)n * HIDDEN + k];
        float v2 = o2[(size_t)n * HIDDEN + k];
        const float* w1 = lw + (size_t)k * N_CLASSES;
        const float* w2 = lw + (size_t)(HIDDEN + k) * N_CLASSES;
        #pragma unroll
        for (int c = 0; c < N_CLASSES; ++c) p[c] += v1 * w1[c] + v2 * w2[c];
    }
    #pragma unroll
    for (int off = 16; off > 0; off >>= 1) {
        #pragma unroll
        for (int c = 0; c < N_CLASSES; ++c) p[c] += __shfl_down(p[c], off, 32);
    }
    if (lane == 0) {
        #pragma unroll
        for (int c = 0; c < N_CLASSES; ++c)
            out[(size_t)n * N_CLASSES + c] = p[c] + lb[c];
    }
}

// ---------------------------------------------------------------- launch
extern "C" void kernel_launch(void* const* d_in, const int* in_sizes, int n_in,
                              void* d_out, int out_size, void* d_ws, size_t ws_size,
                              hipStream_t stream) {
    (void)in_sizes; (void)n_in; (void)out_size; (void)ws_size;

    const float* x     = (const float*)d_in[0];
    const int*   eidx  = (const int*)d_in[1];
    const int*   etype = (const int*)d_in[2];
    const float* W1    = (const float*)d_in[3];
    const float* root1 = (const float*)d_in[4];
    const float* b1    = (const float*)d_in[5];
    const float* W2    = (const float*)d_in[6];
    const float* root2 = (const float*)d_in[7];
    const float* b2    = (const float*)d_in[8];
    const float* lw    = (const float*)d_in[9];
    const float* lb    = (const float*)d_in[10];
    float* out = (float*)d_out;

    const int* src = eidx;            // edge_index[0]
    const int* dst = eidx + N_EDGES;  // edge_index[1]

    // workspace layout: f32 blocks first, then f16 blocks (all 32B aligned)
    float* ws   = (float*)d_ws;
    float* h    = ws;                                     // [N, R, H]   204.8 MB
    float* inv  = h    + (size_t)N_NODES * N_REL * HIDDEN;// [N, R]      1.6 MB
    float* agg  = inv  + (size_t)N_NODES * N_REL;         // [N, H]      25.6 MB
    float* out1 = agg  + (size_t)N_NODES * HIDDEN;        // [N, H]      25.6 MB
    float* out2 = out1 + (size_t)N_NODES * HIDDEN;        // [N, H]      25.6 MB
    _Float16* xh     = (_Float16*)(out2 + (size_t)N_NODES * HIDDEN); // [N,128] f16
    _Float16* Wth    = xh  + (size_t)N_NODES * HIDDEN;    // [R,128,128] f16  256 KB
    _Float16* rootTh = Wth + (size_t)N_REL * N_FEAT * HIDDEN; // [128,128] f16  32 KB

    const int NR = N_NODES * N_REL;          // 400000
    const int NH = N_NODES * HIDDEN;         // 6.4M
    const dim3 gT((N_NODES / 16 + 3) / 4, N_REL);
    const dim3 gR((N_NODES / 16 + 3) / 4);
    const int gE = (N_EDGES + 1) / 2;

    // per-(dst, relation) mean normalization factors (shared by both layers)
    zero_kernel<<<(NR + 255) / 256, 256, 0, stream>>>(inv, NR);
    count_kernel<<<(N_EDGES + 255) / 256, 256, 0, stream>>>(dst, etype, inv);
    inv_kernel<<<(NR + 255) / 256, 256, 0, stream>>>(inv);

    // ---- layer 1: f16 pre-passes, then WMMA GEMMs + edge aggregation
    cast_kernel<<<(NH + 255) / 256, 256, 0, stream>>>(x, xh, NH);
    transpose_cast_kernel<<<dim3(128, N_REL), 128, 0, stream>>>(W1, Wth);
    transpose_cast_kernel<<<dim3(128, 1), 128, 0, stream>>>(root1, rootTh);

    transform_kernel<<<gT, 128, 0, stream>>>(xh, Wth, h);
    zero_kernel<<<(NH + 255) / 256, 256, 0, stream>>>(agg, NH);
    edge_msg_kernel<<<gE, 256, 0, stream>>>(h, src, dst, etype, inv, agg);
    root_relu_kernel<<<gR, 128, 0, stream>>>(xh, rootTh, b1, agg, out1);

    // ---- layer 2
    cast_kernel<<<(NH + 255) / 256, 256, 0, stream>>>(out1, xh, NH);
    transpose_cast_kernel<<<dim3(128, N_REL), 128, 0, stream>>>(W2, Wth);
    transpose_cast_kernel<<<dim3(128, 1), 128, 0, stream>>>(root2, rootTh);

    transform_kernel<<<gT, 128, 0, stream>>>(xh, Wth, h);
    zero_kernel<<<(NH + 255) / 256, 256, 0, stream>>>(agg, NH);
    edge_msg_kernel<<<gE, 256, 0, stream>>>(h, src, dst, etype, inv, agg);
    root_relu_kernel<<<gR, 128, 0, stream>>>(xh, rootTh, b2, agg, out2);

    // ---- classifier head
    final_kernel<<<(N_NODES + 7) / 8, 256, 0, stream>>>(out1, out2, lw, lb, out);
}
